// SinkhornRouter_45329084842149
// MI455X (gfx1250) — matrix operand order repaired and checked by reading, not compiled, not run on hardware
//
#include <hip/hip_runtime.h>
#include <hip/hip_bf16.h>
#include <math.h>

// Problem constants (from reference setup_inputs)
#define Bc 4
#define Nc 4096
#define Dc 1024
#define Ec 8
#define Mc 512   // N / E tokens kept per expert

typedef __attribute__((ext_vector_type(16))) __bf16 v16bf;
typedef __attribute__((ext_vector_type(8)))  __bf16 v8bf;
typedef __attribute__((ext_vector_type(8)))  float  v8f;

__device__ __forceinline__ unsigned short f2bfu(float f) {
    union { float f; unsigned u; } v; v.f = f;
    unsigned r = v.u + 0x7FFFu + ((v.u >> 16) & 1u);  // round-to-nearest-even
    return (unsigned short)(r >> 16);
}
__device__ __forceinline__ __bf16 f2bf(float f) {
    union { unsigned short s; __bf16 b; } o;
    o.s = f2bfu(f);
    return o.b;
}

// LDS byte offset from a generic pointer to __shared__ data: generic shared
// addresses are SHARED_BASE aperture in [63:32], LDS offset in [31:0].
__device__ __forceinline__ unsigned to_lds_off(const void* p) {
    return (unsigned)(unsigned long long)p;
}

// Async copy 16 bytes global -> LDS (no VGPR data round-trip, ASYNCcnt).
__device__ __forceinline__ void async_copy_b128(unsigned lds_off, const void* gaddr) {
    asm volatile("global_load_async_to_lds_b128 %0, %1, off"
                 :: "v"(lds_off), "v"(gaddr) : "memory");
}
__device__ __forceinline__ void wait_async0() {
    asm volatile("s_wait_asynccnt 0x0" ::: "memory");
}

// ---------------------------------------------------------------------------
// Kernel 0: zero the scatter target
// ---------------------------------------------------------------------------
__global__ void zero_kernel(float4* __restrict__ out, int n4) {
    int i = blockIdx.x * blockDim.x + threadIdx.x;
    if (i < n4) out[i] = float4{0.f, 0.f, 0.f, 0.f};
}

// ---------------------------------------------------------------------------
// Kernel 0b: x (B,N,D) f32 -> bf16, same layout
// ---------------------------------------------------------------------------
__global__ void cvt_x(const float4* __restrict__ x, __bf16* __restrict__ xb, int n4) {
    int i = blockIdx.x * blockDim.x + threadIdx.x;
    if (i >= n4) return;
    float4 v = x[i];
    union { unsigned short s[4]; uint2 u; } o;
    o.s[0] = f2bfu(v.x); o.s[1] = f2bfu(v.y);
    o.s[2] = f2bfu(v.z); o.s[3] = f2bfu(v.w);
    *(uint2*)(xb + (size_t)i * 4) = o.u;
}

// ---------------------------------------------------------------------------
// Kernel 0c: experts (E,D,D)[k][f] f32 -> bf16 K-transposed (E,D,D)[f][k]
// 64x64 tile transpose through LDS.
// ---------------------------------------------------------------------------
__global__ __launch_bounds__(256) void cvt_experts(const float* __restrict__ ex,
                                                   __bf16* __restrict__ eb) {
    __shared__ unsigned short t[64][68];  // [f][k], padded
    int blk = blockIdx.x;
    int ft = blk & 15;           // 16 f tiles of 64
    int kt = (blk >> 4) & 15;    // 16 k tiles of 64
    int e  = blk >> 8;
    int tid = threadIdx.x;

    for (int i = tid; i < 64 * 16; i += 256) {
        int kk = i >> 4;
        int f4 = (i & 15) << 2;
        float4 v = *(const float4*)(&ex[((size_t)e * Dc + kt * 64 + kk) * Dc + ft * 64 + f4]);
        t[f4 + 0][kk] = f2bfu(v.x);
        t[f4 + 1][kk] = f2bfu(v.y);
        t[f4 + 2][kk] = f2bfu(v.z);
        t[f4 + 3][kk] = f2bfu(v.w);
    }
    __syncthreads();
    for (int i = tid; i < 64 * 16; i += 256) {
        int f  = i >> 4;
        int k4 = (i & 15) << 2;
        union { unsigned short s[4]; uint2 u; } o;
        o.s[0] = t[f][k4 + 0]; o.s[1] = t[f][k4 + 1];
        o.s[2] = t[f][k4 + 2]; o.s[3] = t[f][k4 + 3];
        *(uint2*)(&eb[((size_t)e * Dc + ft * 64 + f) * Dc + kt * 64 + k4]) = o.u;
    }
}

// ---------------------------------------------------------------------------
// Kernel 1: gate logits -> log(clamp(x @ gate_w, eps)).  One wave per token.
// ---------------------------------------------------------------------------
__global__ __launch_bounds__(256) void gate_kernel(const float* __restrict__ x,
                                                   const float* __restrict__ w,
                                                   float* __restrict__ g) {
    int wid  = (blockIdx.x * blockDim.x + threadIdx.x) >> 5;
    int lane = threadIdx.x & 31;
    if (wid >= Bc * Nc) return;
    const float* xr = x + (size_t)wid * Dc;

    float acc[Ec];
#pragma unroll
    for (int e = 0; e < Ec; ++e) acc[e] = 0.f;

    for (int d0 = lane * 4; d0 < Dc; d0 += 32 * 4) {
        float4 xv = *(const float4*)(xr + d0);
        const float xs[4] = {xv.x, xv.y, xv.z, xv.w};
#pragma unroll
        for (int j = 0; j < 4; ++j) {
            const float4* wr = (const float4*)(w + (size_t)(d0 + j) * Ec);
            float4 w0 = wr[0], w1 = wr[1];
            acc[0] += xs[j] * w0.x; acc[1] += xs[j] * w0.y;
            acc[2] += xs[j] * w0.z; acc[3] += xs[j] * w0.w;
            acc[4] += xs[j] * w1.x; acc[5] += xs[j] * w1.y;
            acc[6] += xs[j] * w1.z; acc[7] += xs[j] * w1.w;
        }
    }
#pragma unroll
    for (int off = 16; off > 0; off >>= 1)
#pragma unroll
        for (int e = 0; e < Ec; ++e) acc[e] += __shfl_xor(acc[e], off, 32);

    if (lane == 0) {
#pragma unroll
        for (int e = 0; e < Ec; ++e)
            g[(size_t)wid * Ec + e] = logf(fmaxf(acc[e], 1e-6f));
    }
}

// ---------------------------------------------------------------------------
// Kernel 2: Sinkhorn column step — logsumexp over tokens n for each (b,e)
// ---------------------------------------------------------------------------
__global__ __launch_bounds__(256) void sink_col(float* __restrict__ g) {
    int be = blockIdx.x;
    int b = be >> 3, e = be & 7;
    __shared__ float red[256];
    int tid = threadIdx.x;

    float m = -1e30f;
    for (int n = tid; n < Nc; n += 256)
        m = fmaxf(m, g[((size_t)b * Nc + n) * Ec + e]);
    red[tid] = m; __syncthreads();
    for (int s = 128; s > 0; s >>= 1) {
        if (tid < s) red[tid] = fmaxf(red[tid], red[tid + s]);
        __syncthreads();
    }
    m = red[0]; __syncthreads();

    float sum = 0.f;
    for (int n = tid; n < Nc; n += 256)
        sum += expf(g[((size_t)b * Nc + n) * Ec + e] - m);
    red[tid] = sum; __syncthreads();
    for (int s = 128; s > 0; s >>= 1) {
        if (tid < s) red[tid] += red[tid + s];
        __syncthreads();
    }
    float lse = m + logf(red[0]);

    for (int n = tid; n < Nc; n += 256)
        g[((size_t)b * Nc + n) * Ec + e] -= lse;
}

// ---------------------------------------------------------------------------
// Kernel 3: Sinkhorn row step — logsumexp over experts e for each (b,n)
// ---------------------------------------------------------------------------
__global__ __launch_bounds__(256) void sink_row(float* __restrict__ g) {
    int t = blockIdx.x * blockDim.x + threadIdx.x;
    if (t >= Bc * Nc) return;
    float v[Ec];
    float m = -1e30f;
#pragma unroll
    for (int e = 0; e < Ec; ++e) { v[e] = g[(size_t)t * Ec + e]; m = fmaxf(m, v[e]); }
    float sum = 0.f;
#pragma unroll
    for (int e = 0; e < Ec; ++e) sum += expf(v[e] - m);
    float lse = m + logf(sum);
#pragma unroll
    for (int e = 0; e < Ec; ++e) g[(size_t)t * Ec + e] = v[e] - lse;
}

// ---------------------------------------------------------------------------
// Kernel 4: top-M=512 tokens per (b,e) by radix-select on monotone u32 keys.
// ---------------------------------------------------------------------------
__global__ __launch_bounds__(256) void topk_kernel(const float* __restrict__ g,
                                                   int* __restrict__ ridx,
                                                   float* __restrict__ gval) {
    int be = blockIdx.x;
    int b = be >> 3, e = be & 7;
    __shared__ unsigned keys[Nc];
    __shared__ int s_cnt;
    __shared__ int s_pos;
    int tid = threadIdx.x;

    for (int n = tid; n < Nc; n += 256) {
        union { float f; unsigned u; } v;
        v.f = g[((size_t)b * Nc + n) * Ec + e];
        keys[n] = (v.u & 0x80000000u) ? ~v.u : (v.u | 0x80000000u);
    }
    __syncthreads();

    unsigned T = 0;
    for (int bit = 31; bit >= 0; --bit) {
        unsigned cand = T | (1u << bit);
        if (tid == 0) s_cnt = 0;
        __syncthreads();
        int c = 0;
        for (int n = tid; n < Nc; n += 256) c += (keys[n] >= cand);
        atomicAdd(&s_cnt, c);
        __syncthreads();
        if (s_cnt >= Mc) T = cand;
        __syncthreads();
    }

    if (tid == 0) s_pos = 0;
    __syncthreads();
    for (int n = tid; n < Nc; n += 256) {
        if (keys[n] > T) {
            int p = atomicAdd(&s_pos, 1);
            int off = ((b * Mc + p) * Ec + e);
            ridx[off] = n;
            gval[off] = expf(g[((size_t)b * Nc + n) * Ec + e]);
        }
    }
    __syncthreads();
    for (int n = tid; n < Nc; n += 256) {
        if (keys[n] == T) {
            int p = atomicAdd(&s_pos, 1);
            if (p < Mc) {
                int off = ((b * Mc + p) * Ec + e);
                ridx[off] = n;
                gval[off] = expf(g[((size_t)b * Nc + n) * Ec + e]);
            }
        }
    }
}

// ---------------------------------------------------------------------------
// Kernel 5a: ASYNC expert GEMM. bf16 inputs pre-staged in workspace; tiles are
// DMA'd memory->LDS with GLOBAL_LOAD_ASYNC_TO_LDS_B128 (double buffered, KT=32),
// WMMA bf16 with f32 accumulate, gate scale, scatter.
// Grid: 1024 = E*B*Mtiles(4)*Ftiles(8); block 256 = 8 waves; tile 128x128.
// ---------------------------------------------------------------------------
__global__ __launch_bounds__(256) void moe_gemm_async(const __bf16* __restrict__ xb,
                                                      const __bf16* __restrict__ eb,
                                                      const int* __restrict__ ridx,
                                                      const float* __restrict__ gval,
                                                      float* __restrict__ out) {
    constexpr int KT = 32;
    constexpr int NCHUNK = Dc / KT;
    __shared__ __bf16 ldsA[2][128][KT];   // [buf][m][k]
    __shared__ __bf16 ldsB[2][128][KT];   // [buf][f][k] (eb already K-transposed)
    __shared__ int    s_tok[128];
    __shared__ float  s_gv[128];

    int blk   = blockIdx.x;
    int ftile = blk & 7;
    int mtile = (blk >> 3) & 3;
    int bIdx  = (blk >> 5) & 3;
    int e     = blk >> 7;
    int tid   = threadIdx.x;
    int m0    = mtile * 128;
    int f0    = ftile * 128;

    if (tid < 128) {
        int off = ((bIdx * Mc + (m0 + tid)) * Ec + e);
        s_tok[tid] = ridx[off];
        s_gv[tid]  = gval[off];
    }
    __syncthreads();

    int lane  = tid & 31;
    int wave  = tid >> 5;
    int waveM = wave & 1;
    int waveF = wave >> 1;

    auto issue = [&](int kc, int buf) {
        // A: 128 rows x 2 chunks of 16B per thread-pass (gathered by token)
        for (int i = tid; i < 128 * 4; i += 256) {
            int r = i >> 2;
            int c = (i & 3) << 3;  // 8 bf16 = 16B
            const __bf16* g = &xb[((size_t)bIdx * Nc + s_tok[r]) * Dc + kc * KT + c];
            async_copy_b128(to_lds_off(&ldsA[buf][r][c]), g);
        }
        // B: 128 f-rows x 2 chunks of 16B per thread-pass
        for (int i = tid; i < 128 * 4; i += 256) {
            int f = i >> 2;
            int c = (i & 3) << 3;
            const __bf16* g = &eb[((size_t)e * Dc + f0 + f) * Dc + kc * KT + c];
            async_copy_b128(to_lds_off(&ldsB[buf][f][c]), g);
        }
    };

    v8f acc[4][2];
    v8f vz = {};
#pragma unroll
    for (int i = 0; i < 4; ++i)
#pragma unroll
        for (int j = 0; j < 2; ++j) acc[i][j] = vz;

    issue(0, 0);

    for (int kc = 0; kc < NCHUNK; ++kc) {
        int cur = kc & 1;
        wait_async0();        // this wave's chunk kc is in LDS
        __syncthreads();      // all waves' chunk kc in LDS; prior buf reads done
        if (kc + 1 < NCHUNK) issue(kc + 1, cur ^ 1);

        int rowSel = lane & 15;
        int klo    = (lane >> 4) << 3;   // A K-halves per ISA layout
        v16bf afrag[4];
#pragma unroll
        for (int ti = 0; ti < 4; ++ti) {
            int row = waveM * 64 + ti * 16 + rowSel;
            v8bf alo = *(v8bf*)(&ldsA[cur][row][klo]);
            v8bf ahi = *(v8bf*)(&ldsA[cur][row][klo + 16]);
            v16bf a;
#pragma unroll
            for (int q = 0; q < 8; ++q) { a[q] = alo[q]; a[q + 8] = ahi[q]; }
            afrag[ti] = a;
        }
        int kb = (lane >> 4) << 3;       // B K layout matches A
        v16bf bfrag[2];
#pragma unroll
        for (int tj = 0; tj < 2; ++tj) {
            int col = waveF * 32 + tj * 16 + rowSel;
            v8bf b0 = *(v8bf*)(&ldsB[cur][col][kb]);
            v8bf b1 = *(v8bf*)(&ldsB[cur][col][kb + 16]);
            v16bf bb;
#pragma unroll
            for (int q = 0; q < 8; ++q) { bb[q] = b0[q]; bb[q + 8] = b1[q]; }
            bfrag[tj] = bb;
        }
#pragma unroll
        for (int ti = 0; ti < 4; ++ti)
#pragma unroll
            for (int tj = 0; tj < 2; ++tj)
                acc[ti][tj] = __builtin_amdgcn_wmma_f32_16x16x32_bf16(
                    false, afrag[ti], false, bfrag[tj],
                    (short)0, acc[ti][tj], false, false);
    }

    int col   = lane & 15;
    int rbase = (lane >> 4) << 3;
#pragma unroll
    for (int ti = 0; ti < 4; ++ti)
#pragma unroll
        for (int tj = 0; tj < 2; ++tj)
#pragma unroll
            for (int j = 0; j < 8; ++j) {
                int mrow = waveM * 64 + ti * 16 + rbase + j;
                int fcol = f0 + waveF * 32 + tj * 16 + col;
                out[((size_t)bIdx * Nc + s_tok[mrow]) * Dc + fcol] =
                    acc[ti][tj][j] * s_gv[mrow];
            }
}

// ---------------------------------------------------------------------------
// Kernel 5b: fallback GEMM (f32 global operands, in-kernel bf16 convert),
// used when workspace is too small for the bf16 mirrors.
// ---------------------------------------------------------------------------
__global__ __launch_bounds__(256) void moe_gemm(const float* __restrict__ x,
                                                const float* __restrict__ experts,
                                                const int* __restrict__ ridx,
                                                const float* __restrict__ gval,
                                                float* __restrict__ out) {
    constexpr int KT = 64;
    __shared__ __bf16 ldsA[128][72];
    __shared__ __bf16 ldsB[128][72];
    __shared__ int    s_tok[128];
    __shared__ float  s_gv[128];

    int blk   = blockIdx.x;
    int ftile = blk & 7;
    int mtile = (blk >> 3) & 3;
    int bIdx  = (blk >> 5) & 3;
    int e     = blk >> 7;
    int tid   = threadIdx.x;
    int m0    = mtile * 128;
    int f0    = ftile * 128;

    if (tid < 128) {
        int off = ((bIdx * Mc + (m0 + tid)) * Ec + e);
        s_tok[tid] = ridx[off];
        s_gv[tid]  = gval[off];
    }
    __syncthreads();

    int lane  = tid & 31;
    int wave  = tid >> 5;
    int waveM = wave & 1;
    int waveF = wave >> 1;

    v8f acc[4][2];
    v8f vz = {};
#pragma unroll
    for (int i = 0; i < 4; ++i)
#pragma unroll
        for (int j = 0; j < 2; ++j) acc[i][j] = vz;

    for (int k0 = 0; k0 < Dc; k0 += KT) {
        if (k0 + KT < Dc) {
            __builtin_prefetch(&experts[((size_t)e * Dc + (k0 + KT) + (tid >> 5)) * Dc + f0 + (tid & 31) * 4], 0, 1);
        }
        for (int i = tid; i < 128 * 16; i += 256) {
            int r  = i >> 4;
            int c4 = (i & 15) << 2;
            float4 v = *(const float4*)(&x[((size_t)bIdx * Nc + s_tok[r]) * Dc + k0 + c4]);
            __bf16* dst = &ldsA[r][c4];
            dst[0] = f2bf(v.x); dst[1] = f2bf(v.y);
            dst[2] = f2bf(v.z); dst[3] = f2bf(v.w);
        }
        for (int i = tid; i < 64 * 32; i += 256) {
            int kk = i >> 5;
            int f4 = (i & 31) << 2;
            float4 v = *(const float4*)(&experts[((size_t)e * Dc + k0 + kk) * Dc + f0 + f4]);
            ldsB[f4 + 0][kk] = f2bf(v.x);
            ldsB[f4 + 1][kk] = f2bf(v.y);
            ldsB[f4 + 2][kk] = f2bf(v.z);
            ldsB[f4 + 3][kk] = f2bf(v.w);
        }
        __syncthreads();

#pragma unroll
        for (int ks = 0; ks < KT; ks += 32) {
            int rowSel = lane & 15;
            int klo    = (lane >> 4) << 3;
            v16bf afrag[4];
#pragma unroll
            for (int ti = 0; ti < 4; ++ti) {
                int row = waveM * 64 + ti * 16 + rowSel;
                v8bf alo = *(v8bf*)(&ldsA[row][ks + klo]);
                v8bf ahi = *(v8bf*)(&ldsA[row][ks + klo + 16]);
                v16bf a;
#pragma unroll
                for (int q = 0; q < 8; ++q) { a[q] = alo[q]; a[q + 8] = ahi[q]; }
                afrag[ti] = a;
            }
            int kb = ks + ((lane >> 4) << 3);
            v16bf bfrag[2];
#pragma unroll
            for (int tj = 0; tj < 2; ++tj) {
                int col = waveF * 32 + tj * 16 + rowSel;
                v8bf b0 = *(v8bf*)(&ldsB[col][kb]);
                v8bf b1 = *(v8bf*)(&ldsB[col][kb + 16]);
                v16bf bb;
#pragma unroll
                for (int q = 0; q < 8; ++q) { bb[q] = b0[q]; bb[q + 8] = b1[q]; }
                bfrag[tj] = bb;
            }
#pragma unroll
            for (int ti = 0; ti < 4; ++ti)
#pragma unroll
                for (int tj = 0; tj < 2; ++tj)
                    acc[ti][tj] = __builtin_amdgcn_wmma_f32_16x16x32_bf16(
                        false, afrag[ti], false, bfrag[tj],
                        (short)0, acc[ti][tj], false, false);
        }
        __syncthreads();
    }

    int col   = lane & 15;
    int rbase = (lane >> 4) << 3;
#pragma unroll
    for (int ti = 0; ti < 4; ++ti)
#pragma unroll
        for (int tj = 0; tj < 2; ++tj)
#pragma unroll
            for (int j = 0; j < 8; ++j) {
                int mrow = waveM * 64 + ti * 16 + rbase + j;
                int fcol = f0 + waveF * 32 + tj * 16 + col;
                out[((size_t)bIdx * Nc + s_tok[mrow]) * Dc + fcol] =
                    acc[ti][tj][j] * s_gv[mrow];
            }
}

// ---------------------------------------------------------------------------
// Host-side launcher
// ---------------------------------------------------------------------------
extern "C" void kernel_launch(void* const* d_in, const int* in_sizes, int n_in,
                              void* d_out, int out_size, void* d_ws, size_t ws_size,
                              hipStream_t stream) {
    (void)in_sizes; (void)n_in; (void)out_size;
    const float* x       = (const float*)d_in[0];   // (B,N,D)
    const float* gate_w  = (const float*)d_in[1];   // (D,E)
    const float* experts = (const float*)d_in[2];   // (E,D,D)
    float* out = (float*)d_out;                     // (B,N,D)

    // Workspace: gates | ridx | gval | [optional bf16 mirrors]
    float* gates = (float*)d_ws;
    int*   ridx  = (int*)(gates + (size_t)Bc * Nc * Ec);
    float* gval  = (float*)(ridx + (size_t)Bc * Mc * Ec);

    size_t base_bytes = (size_t)Bc * Nc * Ec * 4 + (size_t)Bc * Mc * Ec * 8;
    size_t bf_off     = (base_bytes + 255) & ~(size_t)255;
    size_t xb_bytes   = (size_t)Bc * Nc * Dc * 2;
    size_t eb_bytes   = (size_t)Ec * Dc * Dc * 2;
    bool use_async    = ws_size >= bf_off + xb_bytes + eb_bytes;
    __bf16* xb = (__bf16*)((char*)d_ws + bf_off);
    __bf16* eb = (__bf16*)((char*)d_ws + bf_off + xb_bytes);

    // 0) zero scatter target
    {
        int n4 = Bc * Nc * Dc / 4;
        zero_kernel<<<(n4 + 255) / 256, 256, 0, stream>>>((float4*)out, n4);
    }
    // 0b/0c) bf16 mirrors for the async GEMM path
    if (use_async) {
        int n4 = Bc * Nc * Dc / 4;
        cvt_x<<<(n4 + 255) / 256, 256, 0, stream>>>((const float4*)x, xb, n4);
        cvt_experts<<<Ec * 16 * 16, 256, 0, stream>>>(experts, eb);
    }
    // 1) gate logits (log-clamped)
    gate_kernel<<<(Bc * Nc) / 8, 256, 0, stream>>>(x, gate_w, gates);
    // 2) Sinkhorn, 8 iterations in log domain
    for (int it = 0; it < 8; ++it) {
        sink_col<<<Bc * Ec, 256, 0, stream>>>(gates);
        sink_row<<<(Bc * Nc) / 256, 256, 0, stream>>>(gates);
    }
    // 3) top-512 per (b,e)
    topk_kernel<<<Bc * Ec, 256, 0, stream>>>(gates, ridx, gval);
    // 4) expert GEMM + gate + scatter
    if (use_async) {
        moe_gemm_async<<<Ec * Bc * 4 * 8, 256, 0, stream>>>(xb, eb, ridx, gval, out);
    } else {
        moe_gemm<<<Ec * Bc * 4 * 8, 256, 0, stream>>>(x, experts, ridx, gval, out);
    }
}
